// BinarizingLinear_17428977287687
// MI455X (gfx1250) — 8-bit matrix operand order repaired and checked by reading, compiled, not run on hardware
//
#include <hip/hip_runtime.h>
#include <hip/hip_bf16.h>
#include <stdint.h>


typedef __attribute__((ext_vector_type(8))) int v8i;
typedef __attribute__((ext_vector_type(4))) int v4i;
typedef __attribute__((ext_vector_type(2))) int v2i;

#define AS1 __attribute__((address_space(1)))
#define AS3 __attribute__((address_space(3)))

#if defined(__has_builtin)
#if __has_builtin(__builtin_amdgcn_global_load_async_to_lds_b128)
#define HAVE_ASYNC_LDS 1
#endif
#endif

__device__ __forceinline__ void wait_async_copies() {
#if defined(HAVE_ASYNC_LDS)
#if __has_builtin(__builtin_amdgcn_s_wait_asynccnt)
  __builtin_amdgcn_s_wait_asynccnt(0);
#else
  asm volatile("s_wait_asynccnt 0" ::: "memory");
#endif
#endif
}

__global__ __launch_bounds__(256) void binarize_kernel(
    const float* __restrict__ in, int* __restrict__ out, long n4) {
  long i = (long)blockIdx.x * blockDim.x + threadIdx.x;
  if (i >= n4) return;
  const float4 v = ((const float4*)in)[i];
  int b0 = (v.x > 0.f) - (v.x < 0.f);
  int b1 = (v.y > 0.f) - (v.y < 0.f);
  int b2 = (v.z > 0.f) - (v.z < 0.f);
  int b3 = (v.w > 0.f) - (v.w < 0.f);
  out[i] = (b0 & 0xff) | ((b1 & 0xff) << 8) | ((b2 & 0xff) << 16) |
           ((b3 & 0xff) << 24);
}

#define TM 128
#define TN 128
#define TK 64
#define LSTR 80

__global__ __launch_bounds__(256) void bgemm_iu8_kernel(
    const signed char* __restrict__ A,
    const signed char* __restrict__ Bm,
    const float* __restrict__ bias,
    const float* __restrict__ alpha,
    float* __restrict__ out,
    int Mdim, int Ndim, int Kdim) {
  __shared__ __align__(16) signed char ldsA[2][TM * LSTR];
  __shared__ __align__(16) signed char ldsB[2][TM * LSTR];

  const int tid  = threadIdx.x;
  const int lane = tid & 31;
  const int wave = tid >> 5;
  const int wm   = (wave & 3) * 32;
  const int wn   = (wave >> 2) * 64;
  const int m0   = blockIdx.y * TM;
  const int n0   = blockIdx.x * TN;

  const int half = lane >> 4;
  const int mr   = lane & 15;

  auto stage = [&](int buf, int kk) {
#pragma unroll
    for (int j = 0; j < 2; ++j) {
      const int c   = tid + j * 256;
      const int row = c >> 2;
      const int col = (c & 3) * 16;
      const signed char* ga = A + (size_t)(m0 + row) * Kdim + kk + col;
      const signed char* gb = Bm + (size_t)(n0 + row) * Kdim + kk + col;
      signed char* la = &ldsA[buf][row * LSTR + col];
      signed char* lb = &ldsB[buf][row * LSTR + col];
#if defined(HAVE_ASYNC_LDS)
      __builtin_amdgcn_global_load_async_to_lds_b128((v4i*)ga, (AS3 v4i*)la,
                                                     0, 0);
      __builtin_amdgcn_global_load_async_to_lds_b128((v4i*)gb, (AS3 v4i*)lb,
                                                     0, 0);
#else
      *(v4i*)la = *(const v4i*)ga;
      *(v4i*)lb = *(const v4i*)gb;
#endif
    }
  };

  v8i acc[2][4] = {};

  const int nk = Kdim / TK;
  stage(0, 0);

  for (int ks = 0; ks < nk; ++ks) {
    const int buf = ks & 1;
    wait_async_copies();
    __syncthreads();
    if (ks + 1 < nk) stage(buf ^ 1, (ks + 1) * TK);

    const signed char* baseA = ldsA[buf];
    const signed char* baseB = ldsB[buf];

    v8i afrag[2];
#pragma unroll
    for (int t = 0; t < 2; ++t) {
      const signed char* p = baseA + (wm + t * 16 + mr) * LSTR + half * 8;
      v2i q0 = *(const v2i*)(p + 0);
      v2i q1 = *(const v2i*)(p + 16);
      v2i q2 = *(const v2i*)(p + 32);
      v2i q3 = *(const v2i*)(p + 48);
      v8i a = {q0.x, q0.y, q1.x, q1.y, q2.x, q2.y, q3.x, q3.y};
      afrag[t] = a;
    }

    v8i bfrag[4];
#pragma unroll
    for (int t = 0; t < 4; ++t) {
      const signed char* p = baseB + (wn + t * 16 + mr) * LSTR + half * 8;
      v2i p0 = *(const v2i*)(p + 0);
      v2i p1 = *(const v2i*)(p + 16);
      v2i p2 = *(const v2i*)(p + 32);
      v2i p3 = *(const v2i*)(p + 48);
      v8i b = {p0.x, p0.y, p1.x, p1.y, p2.x, p2.y, p3.x, p3.y};
      bfrag[t] = b;
    }

#pragma unroll
    for (int i = 0; i < 2; ++i)
#pragma unroll
      for (int j = 0; j < 4; ++j)
        acc[i][j] = __builtin_amdgcn_wmma_i32_16x16x64_iu8(
             true, afrag[i],  true, bfrag[j], acc[i][j],
             false,  false);

    __syncthreads();
  }

#pragma unroll
  for (int j = 0; j < 4; ++j) {
    const int gn = n0 + wn + j * 16 + mr;
    const float al = alpha[gn];
    const float bi = bias[gn];
#pragma unroll
    for (int i = 0; i < 2; ++i) {
#pragma unroll
      for (int r = 0; r < 8; ++r) {
        const int gm = m0 + wm + i * 16 + half * 8 + r;
        out[(size_t)gm * Ndim + gn] = (float)acc[i][j][r] * al + bi;
      }
    }
  }
}

extern "C" void kernel_launch(void* const* d_in, const int* in_sizes, int n_in,
                              void* d_out, int out_size, void* d_ws,
                              size_t ws_size, hipStream_t stream) {
  (void)in_sizes; (void)n_in; (void)out_size; (void)ws_size;
  const int Bsz = 4, S = 2048, DIN = 4096, DOUT = 4096;
  const int M = Bsz * S;
  const int N = DOUT;
  const int K = DIN;

  const float* x      = (const float*)d_in[0];
  const float* weight = (const float*)d_in[1];
  const float* bias   = (const float*)d_in[2];
  const float* alpha  = (const float*)d_in[3];
  float* out          = (float*)d_out;

  signed char* x_i8 = (signed char*)d_ws;
  signed char* w_i8 = x_i8 + (size_t)M * K;

  {
    long n4 = (long)M * K / 4;
    int blocks = (int)((n4 + 255) / 256);
    binarize_kernel<<<blocks, 256, 0, stream>>>(x, (int*)x_i8, n4);
  }
  {
    long n4 = (long)N * K / 4;
    int blocks = (int)((n4 + 255) / 256);
    binarize_kernel<<<blocks, 256, 0, stream>>>(weight, (int*)w_i8, n4);
  }

  dim3 grid(N / TN, M / TM);
  bgemm_iu8_kernel<<<grid, 256, 0, stream>>>(x_i8, w_i8, bias, alpha, out, M,
                                             N, K);
}
